// RMSCA_36610301231815
// MI455X (gfx1250) — compile-verified, matrix operand order checked by reading, not run on hardware
//
#include <hip/hip_runtime.h>
#include <hip/hip_bf16.h>

typedef _Float16 v16h __attribute__((ext_vector_type(16)));
typedef _Float16 v8h  __attribute__((ext_vector_type(8)));
typedef float    v8f  __attribute__((ext_vector_type(8)));

#define WMMA_F16(a,b,c) __builtin_amdgcn_wmma_f32_16x16x32_f16(false,(a),false,(b),(short)0,(c),false,false)

// ---- problem constants ----
constexpr int C0   = 180;   // channels
constexpr int Kp   = 192;   // padded K / channel dim (6 k-tiles of 32)
constexpr int WIN_T= 900;   // tokens per 30x30 window
constexpr int Mp   = 960;   // padded query rows per window (15 * 64)
constexpr int NW   = 128;   // number of windows (2 * 8 * 8)
constexpr int KV   = 361;   // kv tokens (225 + 100 + 36)
constexpr int KVp  = 384;   // padded kv tokens
constexpr int HD   = 30;    // head dim
constexpr int NH   = 6;     // heads
constexpr int MT   = 57;    // 16-row attention tiles per window (912 >= 900)
constexpr int FRAG = 512;   // halves per fragment record (32 lanes x 16)

// =========================================================================
// fragment-major records: frag = [lane 0..31][16 halves], contiguous.
// f[0..7]  = K (kb..kb+7),  f[8..15] = K (16+kb..16+kb+7), kb = 8*(lane>=16)
// A rows / B cols indexed by lane&15 per CDNA5 ISA 7.12.2.
// =========================================================================
__device__ inline v16h load_frag(const _Float16* p) {
  v8h lo = *(const v8h*)(p);
  v8h hi = *(const v8h*)(p + 8);
  v16h f;
#pragma unroll
  for (int i = 0; i < 8; ++i) { f[i] = lo[i]; f[8 + i] = hi[i]; }
  return f;
}

// A fragment straight from a row-major f16 matrix (contiguous K -> b128 loads)
__device__ inline v16h load_a16(const _Float16* A, int lda, int m0, int k0, int lane) {
  int r  = lane & 15;
  int kb = (lane >> 4) ? 8 : 0;
  const _Float16* p = A + (size_t)(m0 + r) * lda + k0 + kb;
  v8h lo = *(const v8h*)(p);
  v8h hi = *(const v8h*)(p + 16);
  v16h f;
#pragma unroll
  for (int i = 0; i < 8; ++i) { f[i] = lo[i]; f[8 + i] = hi[i]; }
  return f;
}

__device__ inline v16h load_a16_masked(const _Float16* A, int lda, int m0, int k0,
                                       int Mreal, int Kreal, int lane) {
  int r  = lane & 15;
  int kb = (lane >> 4) ? 8 : 0;
  int row = m0 + r;
  bool rok = row < Mreal;
  const _Float16* p = A + (size_t)row * lda + k0 + kb;
  v16h f;
#pragma unroll
  for (int i = 0; i < 8; ++i) {
    int ka = k0 + kb + i;
    int kc = k0 + 16 + kb + i;
    f[i]     = (rok && ka < Kreal) ? p[i]      : (_Float16)0.f;
    f[8 + i] = (rok && kc < Kreal) ? p[16 + i] : (_Float16)0.f;
  }
  return f;
}

// =========================================================================
// kernel 1: weights -> fragment-major f16  [kt(6)][nt(ntiles)][lane][16]
// =========================================================================
__global__ void pack_w_frag(const float* __restrict__ W, int Kreal, int Nreal,
                            int ntiles, _Float16* __restrict__ out) {
  int gid = blockIdx.x * blockDim.x + threadIdx.x;
  int total = 6 * ntiles * 32;
  if (gid >= total) return;
  int lane = gid & 31;
  int frag = gid >> 5;
  int kt = frag / ntiles, nt = frag % ntiles;
  int n  = nt * 16 + (lane & 15);
  int kb = (lane >> 4) ? 8 : 0;
  _Float16* dst = out + (size_t)frag * FRAG + lane * 16;
#pragma unroll
  for (int i = 0; i < 8; ++i) {
    int ka = kt * 32 + kb + i;
    int kc = kt * 32 + 16 + kb + i;
    dst[i]     = (ka < Kreal && n < Nreal) ? (_Float16)W[(size_t)ka * Nreal + n] : (_Float16)0.f;
    dst[8 + i] = (kc < Kreal && n < Nreal) ? (_Float16)W[(size_t)kc * Nreal + n] : (_Float16)0.f;
  }
}

// =========================================================================
// kernel 2: layernorm per window token, gather into [NW][Mp][Kp] f16
// =========================================================================
__global__ void ln_win(const float* __restrict__ x, const float* __restrict__ gamma,
                       const float* __restrict__ beta, _Float16* __restrict__ xn) {
  int wid  = (blockIdx.x * blockDim.x + threadIdx.x) >> 5;
  int lane = threadIdx.x & 31;
  int win = wid / Mp;
  int t   = wid % Mp;
  _Float16* dst = xn + ((size_t)win * Mp + t) * Kp;
  if (t >= WIN_T) {
#pragma unroll
    for (int j = 0; j < 6; ++j) dst[lane + 32 * j] = (_Float16)0.f;
    return;
  }
  int b = win >> 6, wh = (win >> 3) & 7, ww = win & 7;
  int r = t / 30, cc = t % 30;
  size_t base = ((size_t)b * 57600 + (size_t)(wh * 30 + r) * 240 + (ww * 30 + cc)) * 180;
  float xv[6];
  float s = 0.f, s2 = 0.f;
#pragma unroll
  for (int j = 0; j < 6; ++j) {
    int c = lane + 32 * j;
    float v = (c < C0) ? x[base + c] : 0.f;
    xv[j] = v; s += v; s2 += v * v;
  }
#pragma unroll
  for (int m = 16; m; m >>= 1) { s += __shfl_xor(s, m, 32); s2 += __shfl_xor(s2, m, 32); }
  float mu  = s * (1.f / 180.f);
  float var = s2 * (1.f / 180.f) - mu * mu;
  float rstd = rsqrtf(var + 1e-5f);
#pragma unroll
  for (int j = 0; j < 6; ++j) {
    int c = lane + 32 * j;
    float o = (c < C0) ? (xv[j] - mu) * rstd * gamma[c] + beta[c] : 0.f;
    dst[c] = (_Float16)o;
  }
}

// =========================================================================
// kernel 3: multi-scale pooled kv tokens -> [NW][KVp][Kp] f16
// =========================================================================
__global__ void pool_win(const _Float16* __restrict__ xn, _Float16* __restrict__ toks) {
  int wid  = (blockIdx.x * blockDim.x + threadIdx.x) >> 5;
  int lane = threadIdx.x & 31;
  int win = wid / KVp;
  int p   = wid % KVp;
  _Float16* dst = toks + ((size_t)win * KVp + p) * Kp;
  if (p >= KV) {
#pragma unroll
    for (int j = 0; j < 6; ++j) dst[lane + 32 * j] = (_Float16)0.f;
    return;
  }
  int s, gi, gj;
  if (p < 225)      { s = 2; gi = p / 15; gj = p % 15; }
  else if (p < 325) { int q2 = p - 225; s = 3; gi = q2 / 10; gj = q2 % 10; }
  else              { int q2 = p - 325; s = 5; gi = q2 / 6;  gj = q2 % 6;  }
  const _Float16* src = xn + (size_t)win * Mp * Kp;
  float inv = 1.f / (float)(s * s);
#pragma unroll
  for (int j = 0; j < 6; ++j) {
    int c = lane + 32 * j;
    float a = 0.f;
    for (int dy = 0; dy < s; ++dy)
      for (int dx = 0; dx < s; ++dx) {
        int t = (gi * s + dy) * 30 + (gj * s + dx);
        a += (float)src[(size_t)t * Kp + c];
      }
    dst[c] = (_Float16)(a * inv);
  }
}

// =========================================================================
// kernel 4: WMMA GEMM  Out[win] = A[win] @ W + bias  (f16, frag-major weights)
// block = 128 threads (4 waves x 16 rows = 64 rows)
// =========================================================================
__global__ void gemm_f16(const _Float16* __restrict__ A, size_t strideA,
                         const _Float16* __restrict__ Wf, int ntiles,
                         const float* __restrict__ bias, int Nreal,
                         _Float16* __restrict__ Out, size_t strideOut) {
  int win  = blockIdx.y;
  int wave = threadIdx.x >> 5, lane = threadIdx.x & 31;
  int m0 = blockIdx.x * 64 + wave * 16;
  int Np_ = ntiles * 16;
  const _Float16* Aw = A + (size_t)win * strideA;
  _Float16* Ow = Out + (size_t)win * strideOut;
  v16h af[6];
#pragma unroll
  for (int kt = 0; kt < 6; ++kt) af[kt] = load_a16(Aw, Kp, m0, kt * 32, lane);
  for (int nt = 0; nt < ntiles; ++nt) {
    if (nt + 1 < ntiles)
      __builtin_prefetch(Wf + ((size_t)(nt + 1)) * FRAG + lane * 16, 0, 1);
    v8f c = {};
#pragma unroll
    for (int kt = 0; kt < 6; ++kt) {
      v16h bf = load_frag(Wf + ((size_t)(kt * ntiles + nt)) * FRAG + lane * 16);
      c = WMMA_F16(af[kt], bf, c);
    }
    int n  = nt * 16 + (lane & 15);
    int mo = (lane >> 4) ? 8 : 0;
    float bv = (n < Nreal) ? bias[n] : 0.f;
#pragma unroll
    for (int i = 0; i < 8; ++i)
      Ow[(size_t)(m0 + mo + i) * Np_ + n] = (_Float16)(c[i] + bv);
  }
}

// =========================================================================
// kernel 5a: q -> per-head A fragments  [win][h][mt(57)][FRAG]
// =========================================================================
__global__ void repack_q(const _Float16* __restrict__ q, _Float16* __restrict__ qf) {
  int gid = blockIdx.x * blockDim.x + threadIdx.x;
  int total = NW * NH * MT * 32;
  if (gid >= total) return;
  int lane = gid & 31;
  int rec  = gid >> 5;                    // (win*NH + h)*MT + mt
  int mt = rec % MT;
  int wh = rec / MT;
  int h  = wh % NH;
  int win = wh / NH;
  int r  = lane & 15;
  int kb = (lane >> 4) ? 8 : 0;
  const _Float16* p = q + ((size_t)win * Mp + mt * 16 + r) * Kp + h * HD + kb;
  _Float16* dst = qf + (size_t)rec * FRAG + lane * 16;
#pragma unroll
  for (int i = 0; i < 8; ++i) {
    dst[i]     = (kb + i < HD)      ? p[i]      : (_Float16)0.f;
    dst[8 + i] = (16 + kb + i < HD) ? p[16 + i] : (_Float16)0.f;
  }
}

// =========================================================================
// kernel 5b: k -> per-head B fragments (k^T tiles)  [win][h][nt(23)][FRAG]
// =========================================================================
__global__ void repack_k(const _Float16* __restrict__ kv, _Float16* __restrict__ kf) {
  int gid = blockIdx.x * blockDim.x + threadIdx.x;
  int total = NW * NH * 23 * 32;
  if (gid >= total) return;
  int lane = gid & 31;
  int rec  = gid >> 5;                    // (win*NH + h)*23 + nt
  int nt = rec % 23;
  int wh = rec / 23;
  int h  = wh % NH;
  int win = wh / NH;
  int tok = nt * 16 + (lane & 15);
  int kb  = (lane >> 4) ? 8 : 0;
  const _Float16* kp = kv + ((size_t)win * KVp + tok) * KVp + h * HD;
  _Float16* dst = kf + (size_t)rec * FRAG + lane * 16;
#pragma unroll
  for (int i = 0; i < 8; ++i) {
    int ka = kb + i, kc = 16 + kb + i;
    dst[i]     = (ka < HD) ? kp[ka] : (_Float16)0.f;
    dst[8 + i] = (kc < HD) ? kp[kc] : (_Float16)0.f;
  }
}

// =========================================================================
// kernel 5c: v -> per-head B fragments  [win][h][kt(12)][nt2(2)][FRAG]
// =========================================================================
__global__ void repack_v(const _Float16* __restrict__ kv, _Float16* __restrict__ vf) {
  int gid = blockIdx.x * blockDim.x + threadIdx.x;
  int total = NW * NH * 12 * 2 * 32;
  if (gid >= total) return;
  int lane = gid & 31;
  int rec  = gid >> 5;                    // ((win*NH + h)*12 + kt)*2 + nt2
  int nt2 = rec & 1;
  int r2  = rec >> 1;
  int kt = r2 % 12;
  int wh = r2 / 12;
  int h  = wh % NH;
  int win = wh / NH;
  int kb  = (lane >> 4) ? 8 : 0;
  int col = h * HD + nt2 * 16 + (lane & 15);          // <= 181 < 192, in v pad
  const _Float16* vp = kv + ((size_t)win * KVp + kt * 32 + kb) * KVp + Kp + col;
  _Float16* dst = vf + (size_t)rec * FRAG + lane * 16;
#pragma unroll
  for (int i = 0; i < 8; ++i) {
    dst[i]     = vp[(size_t)i * KVp];
    dst[8 + i] = vp[(size_t)(16 + i) * KVp];
  }
}

// =========================================================================
// kernel 6: attention per (window, head, 16-row tile). One wave per block.
//  S = q k^T (23 WMMA), masked softmax over 361 cols, O = P v (24 WMMA)
// =========================================================================
__global__ void attn_kernel(const _Float16* __restrict__ qf,
                            const _Float16* __restrict__ kf,
                            const _Float16* __restrict__ vf,
                            _Float16* __restrict__ o) {
  __shared__ float    S[16][368];
  __shared__ _Float16 P[16][KVp];
  int win = blockIdx.z, h = blockIdx.y;
  int lane = threadIdx.x & 31;
  int mt = blockIdx.x;
  int m0 = mt * 16;
  int wh = win * NH + h;

  v16h aq = load_frag(qf + ((size_t)wh * MT + mt) * FRAG + lane * 16);

  // S = q k^T
  for (int nt = 0; nt < 23; ++nt) {
    if (nt + 1 < 23)
      __builtin_prefetch(kf + ((size_t)wh * 23 + nt + 1) * FRAG + lane * 16, 0, 1);
    v16h bf = load_frag(kf + ((size_t)wh * 23 + nt) * FRAG + lane * 16);
    v8f c = {};
    c = WMMA_F16(aq, bf, c);
    int n  = lane & 15;
    int mo = (lane >> 4) ? 8 : 0;
#pragma unroll
    for (int i = 0; i < 8; ++i) S[mo + i][nt * 16 + n] = c[i];
  }
  __syncthreads();

  // masked softmax (scale = head_dim^-0.5)
  const float scale = 0.18257418583505536f;
  for (int r = 0; r < 16; ++r) {
    float mx = -1e30f;
    for (int col = lane; col < KV; col += 32) mx = fmaxf(mx, S[r][col]);
#pragma unroll
    for (int m = 16; m; m >>= 1) mx = fmaxf(mx, __shfl_xor(mx, m, 32));
    float sum = 0.f;
    for (int col = lane; col < KV; col += 32) {
      float e = __expf((S[r][col] - mx) * scale);
      S[r][col] = e;
      sum += e;
    }
#pragma unroll
    for (int m = 16; m; m >>= 1) sum += __shfl_xor(sum, m, 32);
    float inv = 1.f / sum;
    for (int col = lane; col < KVp; col += 32)
      P[r][col] = (col < KV) ? (_Float16)(S[r][col] * inv) : (_Float16)0.f;
  }
  __syncthreads();

  // O = P @ V   (K = 384 -> 12 k-tiles, N = 32 -> 2 n-tiles)
  for (int nt2 = 0; nt2 < 2; ++nt2) {
    v8f c = {};
    for (int kt = 0; kt < 12; ++kt) {
      v16h ap = load_a16(&P[0][0], KVp, 0, kt * 32, lane);
      v16h bf = load_frag(vf + (((size_t)wh * 12 + kt) * 2 + nt2) * FRAG + lane * 16);
      c = WMMA_F16(ap, bf, c);
    }
    int n  = lane & 15;
    int no = nt2 * 16 + n;
    int mo = (lane >> 4) ? 8 : 0;
    if (no < HD) {
#pragma unroll
      for (int i = 0; i < 8; ++i)
        o[((size_t)win * Mp + m0 + mo + i) * Kp + h * HD + no] = (_Float16)c[i];
    }
  }
}

// =========================================================================
// kernel 7: out = o @ Wp + bp, scatter f32 back into (B, H*W, C)
// masked A loads (rows < 900, k < 180) so unwritten o pads are never read
// =========================================================================
__global__ void gemm_out(const _Float16* __restrict__ A, const _Float16* __restrict__ Wf,
                         const float* __restrict__ bias, float* __restrict__ out) {
  int win  = blockIdx.y;
  int wave = threadIdx.x >> 5, lane = threadIdx.x & 31;
  int m0 = blockIdx.x * 64 + wave * 16;
  const _Float16* Aw = A + (size_t)win * Mp * Kp;
  v16h af[6];
#pragma unroll
  for (int kt = 0; kt < 6; ++kt)
    af[kt] = load_a16_masked(Aw, Kp, m0, kt * 32, WIN_T, C0, lane);
  int b = win >> 6, wh = (win >> 3) & 7, ww = win & 7;
  for (int nt = 0; nt < 12; ++nt) {
    if (nt + 1 < 12)
      __builtin_prefetch(Wf + ((size_t)(nt + 1)) * FRAG + lane * 16, 0, 1);
    v8f c = {};
#pragma unroll
    for (int kt = 0; kt < 6; ++kt) {
      v16h bf = load_frag(Wf + ((size_t)(kt * 12 + nt)) * FRAG + lane * 16);
      c = WMMA_F16(af[kt], bf, c);
    }
    int n  = nt * 16 + (lane & 15);
    int mo = (lane >> 4) ? 8 : 0;
    if (n < C0) {
      float bv = bias[n];
#pragma unroll
      for (int i = 0; i < 8; ++i) {
        int t = m0 + mo + i;
        if (t < WIN_T) {
          int r = t / 30, cc = t % 30;
          size_t idx = ((size_t)b * 57600 + (size_t)(wh * 30 + r) * 240 + (ww * 30 + cc)) * 180 + n;
          out[idx] = c[i] + bv;
        }
      }
    }
  }
}

// =========================================================================
extern "C" void kernel_launch(void* const* d_in, const int* in_sizes, int n_in,
                              void* d_out, int out_size, void* d_ws, size_t ws_size,
                              hipStream_t stream) {
  (void)in_sizes; (void)n_in; (void)out_size; (void)ws_size;
  const float* x     = (const float*)d_in[0];
  const float* Wq    = (const float*)d_in[3];
  const float* bq    = (const float*)d_in[4];
  const float* Wkv   = (const float*)d_in[5];
  const float* bkv   = (const float*)d_in[6];
  const float* Wp    = (const float*)d_in[7];
  const float* bp    = (const float*)d_in[8];
  const float* gamma = (const float*)d_in[9];
  const float* beta  = (const float*)d_in[10];
  float* out = (float*)d_out;

  _Float16* ws   = (_Float16*)d_ws;
  _Float16* WqF  = ws;                                   // 6*12*FRAG
  _Float16* WkvF = WqF  + (size_t)6 * 12 * FRAG;         // 6*24*FRAG
  _Float16* WpF  = WkvF + (size_t)6 * 24 * FRAG;         // 6*12*FRAG
  _Float16* xn   = WpF  + (size_t)6 * 12 * FRAG;         // NW*Mp*Kp
  _Float16* toks = xn   + (size_t)NW * Mp * Kp;          // NW*KVp*Kp
  _Float16* qws  = toks + (size_t)NW * KVp * Kp;         // NW*Mp*Kp
  _Float16* kvws = qws  + (size_t)NW * Mp * Kp;          // NW*KVp*KVp
  _Float16* ows  = kvws + (size_t)NW * KVp * KVp;        // NW*Mp*Kp
  _Float16* vf   = ows  + (size_t)NW * Mp * Kp;          // NW*NH*12*2*FRAG
  // aliases of dead buffers (xn/toks unused after the two projection GEMMs):
  _Float16* qf = xn;    // NW*NH*MT*FRAG = 22.4M halves <= NW*Mp*Kp = 23.6M
  _Float16* kf = toks;  // NW*NH*23*FRAG =  9.0M halves <= NW*KVp*Kp = 9.4M

  pack_w_frag<<<(6 * 12 * 32 + 255) / 256, 256, 0, stream>>>(Wq, 180, 180, 12, WqF);
  pack_w_frag<<<(6 * 24 * 32 + 255) / 256, 256, 0, stream>>>(Wkv, 180, 360, 24, WkvF);
  pack_w_frag<<<(6 * 12 * 32 + 255) / 256, 256, 0, stream>>>(Wp, 180, 180, 12, WpF);

  ln_win<<<NW * Mp / 8, 256, 0, stream>>>(x, gamma, beta, xn);
  pool_win<<<NW * KVp / 8, 256, 0, stream>>>(xn, toks);

  gemm_f16<<<dim3(Mp / 64, NW), 128, 0, stream>>>(xn, (size_t)Mp * Kp, WqF, 12,
                                                  bq, 180, qws, (size_t)Mp * Kp);
  gemm_f16<<<dim3(KVp / 64, NW), 128, 0, stream>>>(toks, (size_t)KVp * Kp, WkvF, 24,
                                                   bkv, 360, kvws, (size_t)KVp * KVp);

  repack_q<<<(NW * NH * MT * 32 + 255) / 256, 256, 0, stream>>>(qws, qf);
  repack_k<<<(NW * NH * 23 * 32 + 255) / 256, 256, 0, stream>>>(kvws, kf);
  repack_v<<<(NW * NH * 12 * 2 * 32 + 255) / 256, 256, 0, stream>>>(kvws, vf);

  attn_kernel<<<dim3(MT, NH, NW), 32, 0, stream>>>(qf, kf, vf, ows);

  gemm_out<<<dim3(Mp / 64, NW), 128, 0, stream>>>(ows, WpF, bp, out);
}
